// Encoder_11957188952075
// MI455X (gfx1250) — compile-verified, matrix operand order checked by reading, not run on hardware
//
#include <hip/hip_runtime.h>
#include <math.h>
#include <stdint.h>

// ---------------------------------------------------------------------------
// MI455X (gfx1250) transformer encoder, fp16 WMMA compute / fp32 accumulate.
// L=4, D=512, H=8 (depth 64), DFF=2048, B=8, S=1024 -> M = 8192 rows.
// ~280 GFLOP, a few hundred MB of traffic at 23.3 TB/s -> matrix-core bound.
// GEMMs: 128x128 block tile, 8 wave32s x (32x64 wave tile) = 8 WMMA/wave/step,
// register double-buffered staging, pair-packed f16 LDS tiles, templated
// branch-free epilogues (f16 output for Q/K/V). Attention: flash-style online
// softmax; K chunks staged via GLOBAL_LOAD_ASYNC_TO_LDS_B128 (ASYNCcnt path),
// Q fragments loaded straight from global f16.
// ---------------------------------------------------------------------------

typedef __attribute__((ext_vector_type(16))) _Float16 v16h;
typedef __attribute__((ext_vector_type(2)))  _Float16 h2;
typedef __attribute__((ext_vector_type(8)))  float    v8f;

#define L_   4
#define D_   512
#define H_   8
#define DEP_ 64
#define DFF_ 2048
#define B_   8
#define S_   1024
#define M_   (B_ * S_)   // 8192

// ---- CDNA5 async global->LDS copy (ASYNCcnt) -------------------------------
// Per-lane: LDS[vdst_lds_byte_addr] = MEM[vaddr], 16B. Tracked by ASYNCcnt.
__device__ __forceinline__ void async_load_b128(unsigned lds_byte_addr,
                                                const void* gaddr) {
  asm volatile("global_load_async_to_lds_b128 %0, %1, off"
               :: "v"(lds_byte_addr), "v"(gaddr)
               : "memory");
}
__device__ __forceinline__ void wait_asynccnt0() {
  asm volatile("s_wait_asynccnt 0" ::: "memory");
}

// ---- WMMA fragment helpers -------------------------------------------------
// 16-bit A-frag (16x32 MxK) per ISA 7.12.2: lane L holds row M=L&15;
// halves j<8 -> K = kb+j, j>=8 -> K = 16+kb+(j-8), kb = (L>>4)*8.
// B-frag (32x16 KxN) mirrors with N = L&15. Each = two 16B loads.

union FragH { v16h h; float4 f[2]; };

__device__ __forceinline__ v16h load_frag(const _Float16* p0, const _Float16* p1) {
  FragH fr;
  fr.f[0] = *(const float4*)p0;
  fr.f[1] = *(const float4*)p1;
  return fr.h;
}

__device__ __forceinline__ h2 pack2(float a, float b) {
  h2 v; v.x = (_Float16)a; v.y = (_Float16)b; return v;
}

// ---- embedding + sinusoidal positional encoding ----------------------------
__global__ __launch_bounds__(256) void embed_kernel(
    const int* __restrict__ x, const float* __restrict__ emb,
    float* __restrict__ h)
{
  const int row = blockIdx.x;            // b*S + s
  const int s   = row & (S_ - 1);
  const int tok = x[row];
  const float scl = sqrtf((float)D_);
  for (int d = threadIdx.x; d < D_; d += 256) {
    float rate = __powf(10000.0f, -(float)(2 * (d >> 1)) / (float)D_);
    float ang  = (float)s * rate;
    float pe   = (d & 1) ? __cosf(ang) : __sinf(ang);
    h[(size_t)row * D_ + d] = emb[(size_t)tok * D_ + d] * scl + pe;
  }
}

// ---- generic GEMM: C = act(A @ W + bias) (+ residual), OutT in {f32,f16} ---
#define BM 128
#define BN 128
#define BK 32
#define LDK (BK + 8)   // padded LDS pitch (halves)

template <int N, int K, bool RELU, bool RESID, typename OutT>
__global__ __launch_bounds__(256) void gemm_kernel(
    const float* __restrict__ A, const float* __restrict__ W,
    const float* __restrict__ bias, const float* __restrict__ residual,
    OutT* __restrict__ C)
{
  __shared__ __align__(16) _Float16 lA[BM][LDK];   // [m][k]
  __shared__ __align__(16) _Float16 lW[BN][LDK];   // [n][k] (transposed)

  const int tid  = threadIdx.x;
  const int m0   = blockIdx.x * BM;
  const int n0   = blockIdx.y * BN;
  const int wv   = tid >> 5, lane = tid & 31;
  const int wm   = wv & 3,   wn   = wv >> 2;       // 4x2 wave grid
  const int lrow = lane & 15, lg = lane >> 4, kb = lg * 8;

  float2 ra[8];   // A: 128x32 = 2048 float2, 8/thread
  float2 rw[8];   // W: 32x128 packed along k, 8/thread

  auto loadA = [&](int k0) {
#pragma unroll
    for (int i = 0; i < 8; ++i) {
      int p = tid + i * 256;
      int r = p >> 4, c2 = p & 15;
      ra[i] = *(const float2*)&A[(size_t)(m0 + r) * K + k0 + 2 * c2];
    }
  };
  auto loadW = [&](int k0) {
#pragma unroll
    for (int i = 0; i < 8; ++i) {
      int p = tid + i * 256;
      int nn = p & 127, kk2 = p >> 7;
      rw[i].x = W[(size_t)(k0 + 2 * kk2)     * N + n0 + nn];
      rw[i].y = W[(size_t)(k0 + 2 * kk2 + 1) * N + n0 + nn];
    }
  };
  auto storeTiles = [&]() {
#pragma unroll
    for (int i = 0; i < 8; ++i) {
      int p = tid + i * 256;
      int r = p >> 4, c2 = p & 15;
      *(h2*)&lA[r][2 * c2] = pack2(ra[i].x, ra[i].y);
    }
#pragma unroll
    for (int i = 0; i < 8; ++i) {
      int p = tid + i * 256;
      int nn = p & 127, kk2 = p >> 7;
      *(h2*)&lW[nn][2 * kk2] = pack2(rw[i].x, rw[i].y);
    }
  };

  v8f acc[2][4] = {};

  loadA(0);
  loadW(0);

  for (int k0 = 0; k0 < K; k0 += BK) {
    __syncthreads();
    storeTiles();
    __syncthreads();

    if (k0 + BK < K) {            // overlap next tile's global loads with WMMA
      loadA(k0 + BK);
      loadW(k0 + BK);
    }
    if (k0 + 2 * BK < K) {        // warm caches two tiles ahead
      __builtin_prefetch(&A[(size_t)(m0 + (tid & 127)) * K + k0 + 2 * BK], 0, 0);
      __builtin_prefetch(&W[(size_t)(k0 + 2 * BK + (tid >> 3)) * N + n0 + (tid & 7) * 16], 0, 0);
    }

    v16h a[2], b[4];
#pragma unroll
    for (int mi = 0; mi < 2; ++mi) {
      int r = wm * 32 + mi * 16 + lrow;
      a[mi] = load_frag(&lA[r][kb], &lA[r][16 + kb]);
    }
#pragma unroll
    for (int nj = 0; nj < 4; ++nj) {
      int cc = wn * 64 + nj * 16 + lrow;
      b[nj] = load_frag(&lW[cc][kb], &lW[cc][16 + kb]);
    }
#pragma unroll
    for (int mi = 0; mi < 2; ++mi)
#pragma unroll
      for (int nj = 0; nj < 4; ++nj)
        acc[mi][nj] = __builtin_amdgcn_wmma_f32_16x16x32_f16(
            false, a[mi], false, b[nj], (short)0, acc[mi][nj], false, false);
  }

  // Branch-free epilogue. C layout: lane holds (row r + 8*lg, col lane&15).
#pragma unroll
  for (int nj = 0; nj < 4; ++nj) {
    const int gcol = n0 + wn * 64 + nj * 16 + lrow;
    const float bv = bias[gcol];
#pragma unroll
    for (int mi = 0; mi < 2; ++mi) {
      const int row0 = m0 + wm * 32 + mi * 16 + 8 * lg;
      OutT* cp = C + (size_t)row0 * N + gcol;
#pragma unroll
      for (int r = 0; r < 8; ++r) {
        float v = acc[mi][nj][r] + bv;
        if (RELU) v = fmaxf(v, 0.0f);
        if constexpr (RESID) {
          const float* rp = residual + (size_t)row0 * N + gcol;
          v += rp[(size_t)r * N];
        }
        cp[(size_t)r * N] = (OutT)v;
      }
    }
  }
}

// ---- flash-style multi-head attention (f16 Q/K/V inputs) -------------------
// One block per (b, h, 128-query chunk). 8 waves x 16 query rows. K chunks
// staged with async global->LDS copies; V transposed through registers.
#define CHK 128

__global__ __launch_bounds__(256) void attn_kernel(
    const _Float16* __restrict__ Qh, const _Float16* __restrict__ Kh,
    const _Float16* __restrict__ Vh, const int* __restrict__ mask,
    float* __restrict__ Out)
{
  __shared__ __align__(16) _Float16 sK [CHK][DEP_ + 8];    // [key][depth]
  __shared__ __align__(16) _Float16 sVT[DEP_][CHK + 8];    // [depth][key]
  __shared__ __align__(16) _Float16 sP [8][16][32 + 8];    // per-wave probs

  const int tid  = threadIdx.x;
  const int wv   = tid >> 5, lane = tid & 31;
  const int lrow = lane & 15, lg = lane >> 4, kb = lg * 8;

  const int bid = blockIdx.x;
  const int qc  = bid & 7;                   // query chunk 0..7
  const int hh  = (bid >> 3) & (H_ - 1);     // head
  const int b   = bid >> 6;                  // batch

  // Q fragments straight from global f16 (two b128 loads each, no cvt).
  const int qrow = qc * 128 + wv * 16 + lrow;
  const _Float16* qp = Qh + ((size_t)(b * S_ + qrow)) * D_ + hh * DEP_;
  const v16h qa0 = load_frag(qp + kb,      qp + 16 + kb);   // depth  0..31
  const v16h qa1 = load_frag(qp + 32 + kb, qp + 48 + kb);   // depth 32..63

  const unsigned sKbase = (unsigned)(uintptr_t)&sK[0][0];

  float m[8], l[8];
  v8f o[4] = {};
#pragma unroll
  for (int r = 0; r < 8; ++r) { m[r] = -1e30f; l[r] = 0.0f; }
  const float scale = 0.125f;                // 1/sqrt(64)

  for (int c0 = 0; c0 < S_; c0 += CHK) {
    __syncthreads();   // all waves done reading previous chunk

    // K chunk: 128 rows x 64 halves, async copied in 16B segments.
#pragma unroll
    for (int i = 0; i < 4; ++i) {
      int p = tid + i * 256;                 // 0..1023 segments
      int kr = p >> 3, seg = p & 7;
      unsigned loff = sKbase + (unsigned)(kr * (DEP_ + 8) + seg * 8) * 2u;
      const void* g = Kh + ((size_t)(b * S_ + c0 + kr)) * D_ + hh * DEP_ + seg * 8;
      async_load_b128(loff, g);
    }
    // V chunk: coalesced b128 f16 loads, transposed b16 stores into sVT.
#pragma unroll
    for (int i = 0; i < 4; ++i) {
      int p = tid + i * 256;
      int kr = p >> 3, seg = p & 7;
      FragH u;
      u.f[0] = *(const float4*)(Vh + ((size_t)(b * S_ + c0 + kr)) * D_ + hh * DEP_ + seg * 8);
#pragma unroll
      for (int j = 0; j < 8; ++j) sVT[seg * 8 + j][kr] = u.h[j];
    }
    wait_asynccnt0();
    __syncthreads();

    for (int jb = 0; jb < CHK; jb += 32) {
      // logits for 32 keys: 2 key-tiles x 2 depth halves = 4 WMMAs
      v8f s0 = {}, s1 = {};
      {
        const _Float16* kp0 = &sK[jb + lrow][0];
        const _Float16* kp1 = &sK[jb + 16 + lrow][0];
        v16h bA0 = load_frag(kp0 + kb,      kp0 + 16 + kb);
        v16h bB0 = load_frag(kp0 + 32 + kb, kp0 + 48 + kb);
        v16h bA1 = load_frag(kp1 + kb,      kp1 + 16 + kb);
        v16h bB1 = load_frag(kp1 + 32 + kb, kp1 + 48 + kb);
        s0 = __builtin_amdgcn_wmma_f32_16x16x32_f16(false, qa0, false, bA0, (short)0, s0, false, false);
        s0 = __builtin_amdgcn_wmma_f32_16x16x32_f16(false, qa1, false, bB0, (short)0, s0, false, false);
        s1 = __builtin_amdgcn_wmma_f32_16x16x32_f16(false, qa0, false, bA1, (short)0, s1, false, false);
        s1 = __builtin_amdgcn_wmma_f32_16x16x32_f16(false, qa1, false, bB1, (short)0, s1, false, false);
      }
      const int keyg = c0 + jb + lrow;
      const float madd0 = (float)mask[b * S_ + keyg]      * -1e9f;
      const float madd1 = (float)mask[b * S_ + keyg + 16] * -1e9f;

      float fac[8];
#pragma unroll
      for (int r = 0; r < 8; ++r) {
        s0[r] = s0[r] * scale + madd0;
        s1[r] = s1[r] * scale + madd1;
        float mx = fmaxf(s0[r], s1[r]);
#pragma unroll
        for (int sh = 8; sh >= 1; sh >>= 1)
          mx = fmaxf(mx, __shfl_xor(mx, sh, 32));
        float mnew = fmaxf(m[r], mx);
        fac[r] = __expf(m[r] - mnew);
        float p0 = __expf(s0[r] - mnew);
        float p1 = __expf(s1[r] - mnew);
        float rs = p0 + p1;
#pragma unroll
        for (int sh = 8; sh >= 1; sh >>= 1)
          rs += __shfl_xor(rs, sh, 32);
        l[r] = l[r] * fac[r] + rs;
        m[r] = mnew;
        sP[wv][r + 8 * lg][lrow]      = (_Float16)p0;
        sP[wv][r + 8 * lg][16 + lrow] = (_Float16)p1;
      }
#pragma unroll
      for (int t = 0; t < 4; ++t)
#pragma unroll
        for (int r = 0; r < 8; ++r) o[t][r] *= fac[r];

      v16h pa = load_frag(&sP[wv][lrow][kb], &sP[wv][lrow][16 + kb]);
#pragma unroll
      for (int t = 0; t < 4; ++t) {
        const _Float16* vp = &sVT[t * 16 + lrow][jb];
        v16h bV = load_frag(vp + kb, vp + 16 + kb);
        o[t] = __builtin_amdgcn_wmma_f32_16x16x32_f16(
            false, pa, false, bV, (short)0, o[t], false, false);
      }
    }
  }

#pragma unroll
  for (int t = 0; t < 4; ++t)
#pragma unroll
    for (int r = 0; r < 8; ++r) {
      int ql = qc * 128 + wv * 16 + r + 8 * lg;
      Out[((size_t)(b * S_ + ql)) * D_ + hh * DEP_ + t * 16 + lrow] =
          o[t][r] / l[r];
    }
}

// ---- layernorm over D=512, one row per block -------------------------------
__global__ __launch_bounds__(256) void layernorm_kernel(
    const float* __restrict__ X, const float* __restrict__ g,
    const float* __restrict__ be, float* __restrict__ Out)
{
  __shared__ float red[40];
  const int row = blockIdx.x;
  const int t = threadIdx.x;
  const float* xp = X + (size_t)row * D_;
  float x0 = xp[t], x1 = xp[t + 256];
  float s = x0 + x1, sq = x0 * x0 + x1 * x1;
#pragma unroll
  for (int sh = 16; sh >= 1; sh >>= 1) {
    s  += __shfl_xor(s, sh, 32);
    sq += __shfl_xor(sq, sh, 32);
  }
  if ((t & 31) == 0) { red[t >> 5] = s; red[8 + (t >> 5)] = sq; }
  __syncthreads();
  if (t < 32) {
    float a  = (t < 8) ? red[t]     : 0.0f;
    float b2 = (t < 8) ? red[8 + t] : 0.0f;
#pragma unroll
    for (int sh = 4; sh >= 1; sh >>= 1) {
      a  += __shfl_xor(a, sh, 32);
      b2 += __shfl_xor(b2, sh, 32);
    }
    if (t == 0) { red[32] = a; red[33] = b2; }
  }
  __syncthreads();
  const float mean = red[32] / (float)D_;
  const float var  = red[33] / (float)D_ - mean * mean;
  const float inv  = rsqrtf(var + 1e-6f);
  Out[(size_t)row * D_ + t]       = (x0 - mean) * inv * g[t]       + be[t];
  Out[(size_t)row * D_ + t + 256] = (x1 - mean) * inv * g[t + 256] + be[t + 256];
}

// ---- orchestration ---------------------------------------------------------
extern "C" void kernel_launch(void* const* d_in, const int* in_sizes, int n_in,
                              void* d_out, int out_size, void* d_ws, size_t ws_size,
                              hipStream_t stream)
{
  const int*   x    = (const int*)  d_in[0];
  const int*   mask = (const int*)  d_in[1];
  const float* emb  = (const float*)d_in[2];
  const float* Wq   = (const float*)d_in[3];  const float* bq  = (const float*)d_in[4];
  const float* Wk   = (const float*)d_in[5];  const float* bk  = (const float*)d_in[6];
  const float* Wv   = (const float*)d_in[7];  const float* bv  = (const float*)d_in[8];
  const float* Wo   = (const float*)d_in[9];  const float* bo  = (const float*)d_in[10];
  const float* W1   = (const float*)d_in[11]; const float* b1  = (const float*)d_in[12];
  const float* W2   = (const float*)d_in[13]; const float* b2  = (const float*)d_in[14];
  const float* g1   = (const float*)d_in[15]; const float* be1 = (const float*)d_in[16];
  const float* g2   = (const float*)d_in[17]; const float* be2 = (const float*)d_in[18];

  char* w = (char*)d_ws;
  float*    h    = (float*)w;    w += (size_t)M_ * D_   * 4;
  _Float16* qh   = (_Float16*)w; w += (size_t)M_ * D_   * 2;
  _Float16* kh   = (_Float16*)w; w += (size_t)M_ * D_   * 2;
  _Float16* vh   = (_Float16*)w; w += (size_t)M_ * D_   * 2;
  float*    attn = (float*)w;    w += (size_t)M_ * D_   * 4;
  float*    out1 = (float*)w;    w += (size_t)M_ * D_   * 4;
  float*    f1   = (float*)w;    w += (size_t)M_ * DFF_ * 4;
  float*    tmp  = (float*)w;

  embed_kernel<<<M_, 256, 0, stream>>>(x, emb, h);

  dim3 g512 (M_ / BM, D_   / BN);   // (64, 4)
  dim3 g2048(M_ / BM, DFF_ / BN);   // (64, 16)

  for (int l = 0; l < L_; ++l) {
    const size_t wdd = (size_t)l * D_ * D_;
    gemm_kernel<D_, D_, false, false, _Float16><<<g512, 256, 0, stream>>>(h, Wq + wdd, bq + l * D_, nullptr, qh);
    gemm_kernel<D_, D_, false, false, _Float16><<<g512, 256, 0, stream>>>(h, Wk + wdd, bk + l * D_, nullptr, kh);
    gemm_kernel<D_, D_, false, false, _Float16><<<g512, 256, 0, stream>>>(h, Wv + wdd, bv + l * D_, nullptr, vh);

    attn_kernel<<<B_ * H_ * (S_ / 128), 256, 0, stream>>>(qh, kh, vh, mask, attn);

    gemm_kernel<D_, D_, false, true, float><<<g512, 256, 0, stream>>>(attn, Wo + wdd, bo + l * D_, h, tmp);
    layernorm_kernel<<<M_, 256, 0, stream>>>(tmp, g1 + l * D_, be1 + l * D_, out1);

    gemm_kernel<DFF_, D_, true, false, float><<<g2048, 256, 0, stream>>>(
        out1, W1 + (size_t)l * D_ * DFF_, b1 + l * DFF_, nullptr, f1);
    gemm_kernel<D_, DFF_, false, true, float><<<g512, 256, 0, stream>>>(
        f1, W2 + (size_t)l * DFF_ * D_, b2 + l * D_, out1, tmp);

    float* dst = (l == L_ - 1) ? (float*)d_out : h;
    layernorm_kernel<<<M_, 256, 0, stream>>>(tmp, g2 + l * D_, be2 + l * D_, dst);
  }
}
